// AV_CrossAttention_13168369729835
// MI455X (gfx1250) — matrix operand order repaired and checked by reading, not compiled, not run on hardware
//
#include <hip/hip_runtime.h>
#include <hip/hip_bf16.h>
#include <math.h>

// AV cross-attention, MI455X (gfx1250, wave32, WMMA).
// N=8192 rows, M=256 features. Strategy: bf16 WMMA recompute (flash-style),
// corr operands (4MB each) stay L2-resident; no 256MB corr materialization.
// Workspace (d_ws) usage: ~20.2 MB.

#define NN 8192
#define MM 256

typedef __attribute__((ext_vector_type(16))) __bf16 v16bf;
typedef __attribute__((ext_vector_type(8)))  float  v8f;

union FragAB { v16bf v; unsigned int u[8]; unsigned short h[16]; };
union FragC  { v8f   v; float f[8]; };

__device__ __forceinline__ unsigned short f32_to_bf16_rne(float x) {
    unsigned int u = __float_as_uint(x);
    unsigned int r = u + 0x7FFFu + ((u >> 16) & 1u);
    return (unsigned short)(r >> 16);
}

// A-fragment (16x32 bf16) from row-major X[ld], tile origin (row0,k0).
// ISA layout: lane l: row=l%16, half=l/16; VGPR0-3: K=half*8+{0..7}; VGPR4-7: K=16+half*8+{0..7}.
__device__ __forceinline__ v16bf load_A_rows(const unsigned short* X, int ld, int row0, int k0) {
    const int lane = threadIdx.x & 31;
    const int r = lane & 15, half = lane >> 4;
    const unsigned short* p = X + (size_t)(row0 + r) * ld + k0 + half * 8;
    const unsigned int* p0 = (const unsigned int*)p;
    const unsigned int* p1 = (const unsigned int*)(p + 16);
    FragAB f;
    f.u[0] = p0[0]; f.u[1] = p0[1]; f.u[2] = p0[2]; f.u[3] = p0[3];
    f.u[4] = p1[0]; f.u[5] = p1[1]; f.u[6] = p1[2]; f.u[7] = p1[3];
    return f.v;
}

// B-fragment (32x16 bf16) where B[k,n] = Y[col0+n, k0+k] (Y row-major, ld stride).
// ISA layout: lane l: col=l%16, half=l/16; VGPR0-3: K=half*8+{0..7}; VGPR4-7: K=16+half*8+{0..7}.
__device__ __forceinline__ v16bf load_B_rows(const unsigned short* Y, size_t ld, int col0, int k0) {
    const int lane = threadIdx.x & 31;
    const int n = lane & 15, kh = lane >> 4;
    const unsigned short* q = Y + (size_t)(col0 + n) * ld + k0 + kh * 8;
    const unsigned int* p0 = (const unsigned int*)q;
    const unsigned int* p1 = (const unsigned int*)(q + 16);
    FragAB f;
    f.u[0] = p0[0]; f.u[1] = p0[1]; f.u[2] = p0[2]; f.u[3] = p0[3];
    f.u[4] = p1[0]; f.u[5] = p1[1]; f.u[6] = p1[2]; f.u[7] = p1[3];
    return f.v;
}

// 16x16 f32 tile of (X @ Y^T) over K=256, both operands row-major bf16 [*,256].
__device__ __forceinline__ v8f corr_tile(const unsigned short* X, const unsigned short* Y,
                                         int row0, int col0) {
    v8f acc = {0.f,0.f,0.f,0.f,0.f,0.f,0.f,0.f};
    #pragma unroll
    for (int k = 0; k < MM; k += 32) {
        v16bf a = load_A_rows(X, MM, row0, k);
        v16bf b = load_B_rows(Y, MM, col0, k);
        acc = __builtin_amdgcn_wmma_f32_16x16x32_bf16(false, a, false, b,
                                                      (short)0, acc, false, false);
    }
    return acc;
}

// ---------------------------------------------------------------------------
// prep: P = audio @ W^T (fp32), then bf16 conversions Pb/Ab/Vb and transposed
// AbT/VbT ([M][N]) for contiguous B-fragment reads in the output GEMMs.
__global__ void prep_kernel(const float* __restrict__ audio, const float* __restrict__ video,
                            const float* __restrict__ W,
                            unsigned short* __restrict__ Pb, unsigned short* __restrict__ Ab,
                            unsigned short* __restrict__ Vb, unsigned short* __restrict__ AbT,
                            unsigned short* __restrict__ VbT) {
    const int idx = blockIdx.x * blockDim.x + threadIdx.x;   // [0, N*M)
    const int i = idx >> 8;
    const int k = idx & 255;
    const float4* a4 = (const float4*)(audio + (size_t)i * MM);
    const float4* w4 = (const float4*)(W + (size_t)k * MM);
    float acc = 0.f;
    #pragma unroll 8
    for (int m = 0; m < MM / 4; ++m) {
        float4 a = a4[m], w = w4[m];
        acc += a.x * w.x + a.y * w.y + a.z * w.z + a.w * w.w;
    }
    Pb[idx] = f32_to_bf16_rne(acc);
    const float av = audio[idx], vv = video[idx];
    Ab[idx] = f32_to_bf16_rne(av);
    Vb[idx] = f32_to_bf16_rne(vv);
    AbT[(size_t)k * NN + i] = f32_to_bf16_rne(av);
    VbT[(size_t)k * NN + i] = f32_to_bf16_rne(vv);
}

// ---------------------------------------------------------------------------
// Row softmax stats of (X @ Y^T): per-row running max & sum(exp) (online).
// Block = 8 waves = 4 row-stripes x 2 column-segments; segment merge in LDS.
__global__ void stats_kernel(const unsigned short* __restrict__ X,
                             const unsigned short* __restrict__ Y,
                             float* __restrict__ omax, float* __restrict__ osum) {
    __shared__ float sm_m[4][16];
    __shared__ float sm_s[4][16];
    const int wave   = threadIdx.x >> 5;
    const int lane   = threadIdx.x & 31;
    const int stripe = wave & 3;
    const int seg    = wave >> 2;
    const int row0   = (blockIdx.x * 4 + stripe) * 16;
    const int rA = lane & 15, half = lane >> 4;

    float m[8], s[8];
    #pragma unroll
    for (int r = 0; r < 8; ++r) { m[r] = -INFINITY; s[r] = 0.f; }

    const int c_beg = seg * (NN / 2);
    for (int c0 = c_beg; c0 < c_beg + NN / 2; c0 += 16) {
        FragC S; S.v = corr_tile(X, Y, row0, c0);
        #pragma unroll
        for (int r = 0; r < 8; ++r) {
            float v = S.f[r];                         // row r+8*half, col c0+rA
            float tm = v;
            #pragma unroll
            for (int off = 1; off < 16; off <<= 1) tm = fmaxf(tm, __shfl_xor(tm, off, 16));
            const float mn = fmaxf(m[r], tm);
            float e = __expf(v - mn);
            #pragma unroll
            for (int off = 1; off < 16; off <<= 1) e += __shfl_xor(e, off, 16);
            s[r] = s[r] * __expf(m[r] - mn) + e;
            m[r] = mn;
        }
    }
    if (seg == 1 && rA == 0) {
        #pragma unroll
        for (int r = 0; r < 8; ++r) { sm_m[stripe][r + 8*half] = m[r]; sm_s[stripe][r + 8*half] = s[r]; }
    }
    __syncthreads();
    if (seg == 0 && rA == 0) {
        #pragma unroll
        for (int r = 0; r < 8; ++r) {
            const int row = r + 8 * half;
            const float m1 = sm_m[stripe][row], s1 = sm_s[stripe][row];
            const float mg = fmaxf(m[r], m1);
            const float sg = s[r] * __expf(m[r] - mg) + s1 * __expf(m1 - mg);
            omax[row0 + row] = mg;
            osum[row0 + row] = sg;
        }
    }
}

// ---------------------------------------------------------------------------
// out[row0..row0+15, outoff..outoff+255] = tanh( sum_c E[row,c]*V[c,:] + resid[row,:] )
// where E = exp((Xrows@Ycols^T) - smax[col]) / ssum[col].
// VT is the value matrix transposed ([M][N], bf16) so B-fragments read contiguously.
// Block = 8 waves = 4 stripes x 2 column-segments; partial accumulators combined in LDS.
__global__ void attn_out_kernel(const unsigned short* __restrict__ Xrows,
                                const unsigned short* __restrict__ Ycols,
                                const float* __restrict__ smax, const float* __restrict__ ssum,
                                const unsigned short* __restrict__ VT,
                                const float* __restrict__ resid,
                                float* __restrict__ out, int outld, int outoff) {
    __shared__ unsigned short wbuf[8][16 * 32];   // per-wave 16x32 bf16 weight tile
    __shared__ float cbuf[4][16][16];             // segment-combine chunks
    const int wave   = threadIdx.x >> 5;
    const int lane   = threadIdx.x & 31;
    const int stripe = wave & 3;
    const int seg    = wave >> 2;
    const int row0   = (blockIdx.x * 4 + stripe) * 16;
    const int rA = lane & 15, half = lane >> 4;
    unsigned short* myw = &wbuf[wave][0];

    v8f acc[16];
    #pragma unroll
    for (int t = 0; t < 16; ++t) acc[t] = (v8f){0.f,0.f,0.f,0.f,0.f,0.f,0.f,0.f};

    const int c_beg = seg * (NN / 2);
    for (int c0 = c_beg; c0 < c_beg + NN / 2; c0 += 32) {
        // --- weights: two 16x16 corr tiles, softmax-normalized per column ---
        #pragma unroll
        for (int t = 0; t < 2; ++t) {
            const int cc = c0 + t * 16;
            FragC S; S.v = corr_tile(Xrows, Ycols, row0, cc);
            const float mx  = smax[cc + rA];
            const float inv = 1.0f / ssum[cc + rA];
            #pragma unroll
            for (int r = 0; r < 8; ++r) {
                const float w = __expf(S.f[r] - mx) * inv;
                myw[(r + 8 * half) * 32 + t * 16 + rA] = f32_to_bf16_rne(w);
            }
        }
        __builtin_amdgcn_wave_barrier();   // same-wave DS ops are in-order; fence compiler
        // --- reload weights in A-fragment layout (lane-transpose via LDS) ---
        FragAB af;
        {
            const unsigned short* base = myw + rA * 32;
            const unsigned int* lo = (const unsigned int*)(base + half * 8);
            const unsigned int* hi = (const unsigned int*)(base + 16 + half * 8);
            af.u[0] = lo[0]; af.u[1] = lo[1]; af.u[2] = lo[2]; af.u[3] = lo[3];
            af.u[4] = hi[0]; af.u[5] = hi[1]; af.u[6] = hi[2]; af.u[7] = hi[3];
        }
        __builtin_amdgcn_wave_barrier();
        // --- acc(16x256) += W(16x32) @ V(32x256), B[k,n] = VT[(t*16+n)*N + c0+k] ---
        #pragma unroll
        for (int t = 0; t < 16; ++t) {
            FragAB bf;
            const unsigned int* p =
                (const unsigned int*)(VT + (size_t)(t * 16 + rA) * NN + c0 + half * 8);
            #pragma unroll
            for (int i = 0; i < 8; ++i) bf.u[i] = p[(i & 3) + (i >> 2) * 8];
            acc[t] = __builtin_amdgcn_wmma_f32_16x16x32_bf16(false, af.v, false, bf.v,
                                                             (short)0, acc[t], false, false);
        }
    }

    // --- combine the two column-segments (deterministic, chunked through LDS) ---
    __syncthreads();
    for (int t = 0; t < 16; ++t) {
        if (seg == 1) {
            FragC c; c.v = acc[t];
            #pragma unroll
            for (int r = 0; r < 8; ++r) cbuf[stripe][r + 8 * half][rA] = c.f[r];
        }
        __syncthreads();
        if (seg == 0) {
            FragC c; c.v = acc[t];
            #pragma unroll
            for (int r = 0; r < 8; ++r) c.f[r] += cbuf[stripe][r + 8 * half][rA];
            acc[t] = c.v;
        }
        __syncthreads();
    }

    // --- epilogue: tanh(acc + resid), write concat half ---
    if (seg == 0) {
        #pragma unroll
        for (int t = 0; t < 16; ++t) {
            FragC c; c.v = acc[t];
            #pragma unroll
            for (int r = 0; r < 8; ++r) {
                const int row = row0 + r + 8 * half;
                const int col = t * 16 + rA;
                const float v = tanhf(c.f[r] + resid[(size_t)row * MM + col]);
                out[(size_t)row * outld + outoff + col] = v;
            }
        }
    }
}

// ---------------------------------------------------------------------------
extern "C" void kernel_launch(void* const* d_in, const int* in_sizes, int n_in,
                              void* d_out, int out_size, void* d_ws, size_t ws_size,
                              hipStream_t stream) {
    (void)in_sizes; (void)n_in; (void)out_size; (void)ws_size;
    const float* audio = (const float*)d_in[0];
    const float* video = (const float*)d_in[1];
    const float* W     = (const float*)d_in[2];
    float* out = (float*)d_out;

    char* ws = (char*)d_ws;
    const size_t MB = 1024 * 1024;
    unsigned short* Pb  = (unsigned short*)(ws);              // [N][M] bf16, 4MB
    unsigned short* Ab  = (unsigned short*)(ws + 4  * MB);    // [N][M] bf16
    unsigned short* Vb  = (unsigned short*)(ws + 8  * MB);    // [N][M] bf16
    unsigned short* AbT = (unsigned short*)(ws + 12 * MB);    // [M][N] bf16
    unsigned short* VbT = (unsigned short*)(ws + 16 * MB);    // [M][N] bf16
    float* rmax = (float*)(ws + 20 * MB);
    float* rsum = rmax + NN;
    float* cmax = rsum + NN;
    float* csum = cmax + NN;

    prep_kernel<<<(NN * MM) / 256, 256, 0, stream>>>(audio, video, W, Pb, Ab, Vb, AbT, VbT);

    // Row stats of corr = Pb @ Vb^T  -> rmax/rsum
    stats_kernel<<<NN / 64, 256, 0, stream>>>(Pb, Vb, rmax, rsum);
    // Col stats of corr = row stats of corr^T = Vb @ Pb^T -> cmax/csum
    stats_kernel<<<NN / 64, 256, 0, stream>>>(Vb, Pb, cmax, csum);

    // d_video[j,:] = sum_i exp(corr[i,j]-rmax[i])/rsum[i] * video[i,:]  (tiles of corr^T)
    attn_out_kernel<<<NN / 64, 256, 0, stream>>>(Vb, Pb, rmax, rsum, VbT, video, out, 2 * MM, 0);
    // d_audio[i,:] = sum_j exp(corr[i,j]-cmax[j])/csum[j] * audio[j,:]
    attn_out_kernel<<<NN / 64, 256, 0, stream>>>(Pb, Vb, cmax, csum, AbT, audio, out, 2 * MM, MM);
}
